// SparseAttention_47502338294473
// MI455X (gfx1250) — matrix operand order repaired and checked by reading, not compiled, not run on hardware
//
#include <hip/hip_runtime.h>

// ---------------------------------------------------------------------------
// Sparse top-k attention for MI455X (gfx1250), wave32 + WMMA f32_16x16x32_f16,
// double-buffered async global->LDS staging (ASYNCcnt pipelining) for GEMMs.
// ---------------------------------------------------------------------------
#define D_MODEL   1024
#define NUM_HEADS 16
#define TOPK      8
#define DK        64
#define SEQ       2048
#define BATCH     2
#define BS        (BATCH * SEQ)     // 4096 rows
#define LN_EPS    1e-5f
#define NEG_BIG   (-3.0e38f)
#define STRIP     512               // score columns per LDS strip

// GEMM block tiling
#define BM   128
#define BN   128
#define BK   32
#define LDSP 40                     // padded halves per LDS row (80 B, 16B-aligned)

typedef __attribute__((ext_vector_type(16))) _Float16 v16h;
typedef __attribute__((ext_vector_type(8)))  _Float16 v8h;
typedef __attribute__((ext_vector_type(4)))  _Float16 v4h;
typedef __attribute__((ext_vector_type(8)))  float    v8f;
typedef __attribute__((ext_vector_type(4)))  float    v4f;

#define WMMA_F16(a, b, c) \
  __builtin_amdgcn_wmma_f32_16x16x32_f16(false, (a), false, (b), (short)0, (c), false, false)

// ---------------------------------------------------------------------------
// CDNA5 async global -> LDS copy (16 B per lane per instruction; ASYNCcnt).
// ---------------------------------------------------------------------------
__device__ __forceinline__ void async_copy_b128(unsigned lds_addr, const void* gaddr) {
  asm volatile("global_load_async_to_lds_b128 %0, %1, off"
               :: "v"(lds_addr), "v"((unsigned long long)(uintptr_t)gaddr)
               : "memory");
}
// Wait until at most `n` async ops remain outstanding (in-order completion).
#define WAIT_ASYNC(n) asm volatile("s_wait_asynccnt " #n ::: "memory")

// 16x32 f16 A-tile (M x K), row-major with leading dim `ld` (halves).
// ISA layout: lanes 0-15 hold K={0..7,16..23}, lanes 16-31 hold K={8..15,24..31}.
__device__ __forceinline__ v16h load_a16x32(const _Float16* base, int ld,
                                            int m0, int k0, int lane) {
  const int r  = lane & 15;
  const int hi = lane >> 4;
  const _Float16* p = base + (size_t)(m0 + r) * ld + k0 + hi * 8;
  v8h lo = *(const v8h*)(p);
  v8h hh = *(const v8h*)(p + 16);
  v16h out;
#pragma unroll
  for (int i = 0; i < 8; ++i) { out[i] = lo[i]; out[i + 8] = hh[i]; }
  return out;
}

// 32x16 f16 B-tile where B[k][n] = W[n0+n][k0+k], W row-major with leading dim ld.
// ISA layout: lanes 0-15 -> col n with K={0..7,16..23}, lanes 16-31 -> same cols K={8..15,24..31}.
__device__ __forceinline__ v16h load_bT32x16(const _Float16* base, int ld,
                                             int n0, int k0, int lane) {
  const int n  = lane & 15;
  const int hi = lane >> 4;
  const _Float16* p = base + (size_t)(n0 + n) * ld + k0 + hi * 8;
  v8h x0 = *(const v8h*)(p);
  v8h x1 = *(const v8h*)(p + 16);
  v16h out;
#pragma unroll
  for (int i = 0; i < 8; ++i) { out[i] = x0[i]; out[i + 8] = x1[i]; }
  return out;
}

// ---------------------------------------------------------------------------
// Kernel 1: f32 -> f16 conversion, vectorized (b128 in / b64 out)
// ---------------------------------------------------------------------------
__global__ void k_f32_to_f16(const float* __restrict__ src,
                             _Float16* __restrict__ dst, int n) {
  int i = (blockIdx.x * blockDim.x + threadIdx.x) * 4;
  const int stride = gridDim.x * blockDim.x * 4;
  for (; i < n; i += stride) {
    v4f x = *(const v4f*)(src + i);
    v4h y;
#pragma unroll
    for (int j = 0; j < 4; ++j) y[j] = (_Float16)x[j];
    *(v4h*)(dst + i) = y;
  }
}

// ---------------------------------------------------------------------------
// Kernel 2: tiled GEMM  C[M,N] = X[M,K] @ W[N,K]^T, double-buffered async LDS.
// 256 threads = 8 waves; block tile 128x128; wave tile 64x32 (8 WMMA accums).
// mode 0: f16 out, [B,H,S,DK] head layout (Q,K)
// mode 1: f32 out, [B,H,S,DK] head layout (V)
// mode 2: f32 out, flat [BS,D] with +bias +residual (out-projection)
// ---------------------------------------------------------------------------
__global__ __launch_bounds__(256) void k_gemm_xwT(
    const _Float16* __restrict__ xh, const _Float16* __restrict__ wh,
    _Float16* __restrict__ o16, float* __restrict__ o32,
    const float* __restrict__ bias, const float* __restrict__ residual,
    int mode) {
  __shared__ _Float16 As[2][BM * LDSP];   // 2 x 10 KB
  __shared__ _Float16 Bs[2][BN * LDSP];   // 2 x 10 KB

  const int lane = threadIdx.x & 31;
  const int wave = threadIdx.x >> 5;   // 0..7
  const int wm = wave >> 2;            // 0..1  -> 64 rows each
  const int wn = wave & 3;             // 0..3  -> 32 cols each
  const int bm0 = blockIdx.y * BM;
  const int bn0 = blockIdx.x * BN;

  // Per-thread staging coordinates: 512 16B-chunks per panel, 2 per thread.
  const int row0 = threadIdx.x >> 2;             // rows 0..63
  const int row1 = (threadIdx.x + 256) >> 2;     // rows 64..127
  const int cc0  = threadIdx.x & 3;
  const unsigned loff0 = (unsigned)(row0 * LDSP + cc0 * 8) * 2;
  const unsigned loff1 = (unsigned)(row1 * LDSP + cc0 * 8) * 2;

  // Issues 4 async b128 copies (2 A + 2 B) for panel at k0 into buffer `buf`.
  auto stage = [&](int buf, int k0) {
    const unsigned a0 = (unsigned)(uintptr_t)(void*)&As[buf][0];
    const unsigned b0 = (unsigned)(uintptr_t)(void*)&Bs[buf][0];
    async_copy_b128(a0 + loff0, xh + (size_t)(bm0 + row0) * D_MODEL + k0 + cc0 * 8);
    async_copy_b128(b0 + loff0, wh + (size_t)(bn0 + row0) * D_MODEL + k0 + cc0 * 8);
    async_copy_b128(a0 + loff1, xh + (size_t)(bm0 + row1) * D_MODEL + k0 + cc0 * 8);
    async_copy_b128(b0 + loff1, wh + (size_t)(bn0 + row1) * D_MODEL + k0 + cc0 * 8);
  };

  v8f acc[4][2] = {};

  stage(0, 0);                               // prologue prefetch
#pragma unroll 1
  for (int k0 = 0; k0 < D_MODEL; k0 += BK) {
    const int cur = (k0 / BK) & 1;
    if (k0 + BK < D_MODEL) {
      stage(cur ^ 1, k0 + BK);               // prefetch next panel (stays in flight)
      WAIT_ASYNC(4);                         // in-order: current panel complete
    } else {
      WAIT_ASYNC(0);
    }
    __syncthreads();

    // --- compute from LDS: 8 WMMA per wave per K-step ---
    const _Float16* Ac = As[cur];
    const _Float16* Bc = Bs[cur];
    v16h bfrag[2];
#pragma unroll
    for (int j = 0; j < 2; ++j)
      bfrag[j] = load_bT32x16(Bc, LDSP, wn * 32 + j * 16, 0, lane);
#pragma unroll
    for (int i = 0; i < 4; ++i) {
      v16h afrag = load_a16x32(Ac, LDSP, wm * 64 + i * 16, 0, lane);
#pragma unroll
      for (int j = 0; j < 2; ++j)
        acc[i][j] = WMMA_F16(afrag, bfrag[j], acc[i][j]);
    }
    __syncthreads();   // reads done before this buffer is restaged next+1 iter
  }

  // --- write out 8 16x16 tiles per wave ---
  const int n  = lane & 15;
  const int hi = lane >> 4;
#pragma unroll
  for (int i = 0; i < 4; ++i) {
#pragma unroll
    for (int j = 0; j < 2; ++j) {
      const int m0t = bm0 + wm * 64 + i * 16;
      const int col = bn0 + wn * 32 + j * 16 + n;
      if (mode == 2) {
        const float bv = bias[col];
#pragma unroll
        for (int r = 0; r < 8; ++r) {
          const int m = m0t + r + 8 * hi;
          const size_t o = (size_t)m * D_MODEL + col;
          o32[o] = acc[i][j][r] + bv + residual[o];
        }
      } else {
        const int h = col >> 6;
        const int d = col & 63;
#pragma unroll
        for (int r = 0; r < 8; ++r) {
          const int m  = m0t + r + 8 * hi;
          const int b_ = m >> 11;
          const int s  = m & (SEQ - 1);
          const size_t idx = (((size_t)(b_ * NUM_HEADS + h) * SEQ) + s) * DK + d;
          if (mode == 0) o16[idx] = (_Float16)acc[i][j][r];
          else           o32[idx] = acc[i][j][r];
        }
      }
    }
  }
}

// ---------------------------------------------------------------------------
// Kernel 3: fused scores (WMMA) + streaming top-8 + softmax + V gather.
// One 2-wave block per (b, h, 16 query rows). LDS strip of 16x512 scores.
// ---------------------------------------------------------------------------
__global__ __launch_bounds__(64) void k_scores_topk(
    const _Float16* __restrict__ Qh,   // [B,H,S,DK] f16
    const _Float16* __restrict__ Kh,   // [B,H,S,DK] f16
    const float*    __restrict__ Vf,   // [B,H,S,DK] f32
    _Float16* __restrict__ attn_out,   // [B,S,D]   f16
    float* __restrict__ sw_out) {      // [B,H,S,TOPK] f32
  __shared__ float sc[16 * STRIP];     // 32 KB score strip
  __shared__ float s_tv[16][TOPK];     // persistent per-row top-8 values
  __shared__ int   s_ti[16][TOPK];     // persistent per-row top-8 indices

  const int lane = threadIdx.x & 31;
  const int wave = threadIdx.x >> 5;   // 0..1
  const int blk  = blockIdx.x;         // B*H*(S/16) = 4096
  const int qt = blk & 127;
  const int h  = (blk >> 7) & (NUM_HEADS - 1);
  const int b  = blk >> 11;
  const int q0 = qt * 16;

  const _Float16* Qbase = Qh + (((size_t)(b * NUM_HEADS + h) * SEQ) + q0) * DK;
  const _Float16* Kbase = Kh + ((size_t)(b * NUM_HEADS + h) * SEQ) * DK;
  const float*    Vbase = Vf + ((size_t)(b * NUM_HEADS + h) * SEQ) * DK;

  if (threadIdx.x < 16) {
#pragma unroll
    for (int t = 0; t < TOPK; ++t) { s_tv[threadIdx.x][t] = NEG_BIG; s_ti[threadIdx.x][t] = 0; }
  }

  // Q tile 16x64 resident in registers for the whole kernel.
  v16h a0 = load_a16x32(Qbase, DK, 0, 0, lane);
  v16h a1 = load_a16x32(Qbase, DK, 0, 32, lane);
  __syncthreads();

  for (int strip = 0; strip < SEQ / STRIP; ++strip) {
    const int kbase = strip * STRIP;
    // --- score strip via WMMA (waves split key tiles) ---
    for (int kt = wave; kt < STRIP / 16; kt += 2) {
      const int key0 = kbase + kt * 16;
      v16h b0 = load_bT32x16(Kbase, DK, key0, 0, lane);
      v16h b1 = load_bT32x16(Kbase, DK, key0, 32, lane);
      v8f c = {};
      c = WMMA_F16(a0, b0, c);
      c = WMMA_F16(a1, b1, c);
      const int n  = lane & 15;
      const int hi = lane >> 4;
#pragma unroll
      for (int r = 0; r < 8; ++r)
        sc[(r + 8 * hi) * STRIP + kt * 16 + n] = c[r] * 0.125f;  // 1/sqrt(64)
    }
    __syncthreads();

    // --- strip top-8 per row, merged into persistent sorted list ---
    for (int rr = 0; rr < 8; ++rr) {
      const int row = wave * 8 + rr;
      float nv[TOPK]; int ni[TOPK];
      for (int t = 0; t < TOPK; ++t) {
        float best = NEG_BIG; int bidx = -1;
        for (int j = lane; j < STRIP; j += 32) {
          const float v = sc[row * STRIP + j];
          if (v > best) { best = v; bidx = j; }
        }
#pragma unroll
        for (int off = 16; off > 0; off >>= 1) {
          const float ov = __shfl_xor(best, off, 32);
          const int   oi = __shfl_xor(bidx, off, 32);
          if (ov > best || (ov == best && oi >= 0 && (bidx < 0 || oi < bidx))) {
            best = ov; bidx = oi;
          }
        }
        nv[t] = best; ni[t] = bidx + kbase;
        if (lane == 0 && bidx >= 0) sc[row * STRIP + bidx] = NEG_BIG;
      }
      // merge two descending-sorted 8-lists (old indices are smaller: ties -> old)
      float cv[TOPK]; int ci[TOPK];
#pragma unroll
      for (int t = 0; t < TOPK; ++t) { cv[t] = s_tv[row][t]; ci[t] = s_ti[row][t]; }
      float mv[TOPK]; int mi[TOPK];
      int ia = 0, ib = 0;
#pragma unroll
      for (int t = 0; t < TOPK; ++t) {
        const bool takeA = (ib >= TOPK) || (ia < TOPK && cv[ia] >= nv[ib]);
        if (takeA) { mv[t] = cv[ia]; mi[t] = ci[ia]; ++ia; }
        else       { mv[t] = nv[ib]; mi[t] = ni[ib]; ++ib; }
      }
      if (lane == 0) {
#pragma unroll
        for (int t = 0; t < TOPK; ++t) { s_tv[row][t] = mv[t]; s_ti[row][t] = mi[t]; }
      }
    }
    __syncthreads();
  }

  // --- softmax over top-8 + gather-weighted sum of V ---
  for (int rr = 0; rr < 8; ++rr) {
    const int row = wave * 8 + rr;
    const int q = q0 + row;
    float w[TOPK]; int idx[TOPK];
#pragma unroll
    for (int t = 0; t < TOPK; ++t) { w[t] = s_tv[row][t]; idx[t] = s_ti[row][t]; }
    const float mx = w[0];                  // list is sorted descending
    float sum = 0.f;
#pragma unroll
    for (int t = 0; t < TOPK; ++t) { w[t] = __expf(w[t] - mx); sum += w[t]; }
    const float inv = 1.0f / sum;
#pragma unroll
    for (int t = 0; t < TOPK; ++t) w[t] *= inv;

    if (lane < TOPK)
      sw_out[(((size_t)(b * NUM_HEADS + h) * SEQ) + q) * TOPK + lane] = w[lane];

    float acc0 = 0.f, acc1 = 0.f;
#pragma unroll
    for (int t = 0; t < TOPK; ++t) {
      const float* vrow = Vbase + (size_t)idx[t] * DK;
      acc0 += w[t] * vrow[lane];
      acc1 += w[t] * vrow[lane + 32];
    }
    _Float16* dst = attn_out + ((size_t)(b * SEQ + q)) * D_MODEL + h * DK;
    dst[lane]      = (_Float16)acc0;
    dst[lane + 32] = (_Float16)acc1;
  }
}

// ---------------------------------------------------------------------------
// Kernel 4: row LayerNorm over D=1024, one 256-thread block per row
// ---------------------------------------------------------------------------
__global__ __launch_bounds__(256) void k_layernorm(
    const float* __restrict__ x, const float* __restrict__ gamma,
    const float* __restrict__ beta, float* __restrict__ out) {
  __shared__ float red[8];
  __shared__ float s_mu, s_rstd;
  const int row = blockIdx.x;
  const float* xr = x + (size_t)row * D_MODEL;

  float s = 0.f;
  for (int j = threadIdx.x; j < D_MODEL; j += 256) s += xr[j];
#pragma unroll
  for (int off = 16; off > 0; off >>= 1) s += __shfl_xor(s, off, 32);
  if ((threadIdx.x & 31) == 0) red[threadIdx.x >> 5] = s;
  __syncthreads();
  if (threadIdx.x == 0) {
    float t = 0.f;
    for (int i = 0; i < 8; ++i) t += red[i];
    s_mu = t * (1.0f / D_MODEL);
  }
  __syncthreads();
  const float mu = s_mu;

  float v = 0.f;
  for (int j = threadIdx.x; j < D_MODEL; j += 256) {
    const float d = xr[j] - mu;
    v += d * d;
  }
#pragma unroll
  for (int off = 16; off > 0; off >>= 1) v += __shfl_xor(v, off, 32);
  if ((threadIdx.x & 31) == 0) red[threadIdx.x >> 5] = v;
  __syncthreads();
  if (threadIdx.x == 0) {
    float t = 0.f;
    for (int i = 0; i < 8; ++i) t += red[i];
    s_rstd = rsqrtf(t * (1.0f / D_MODEL) + LN_EPS);
  }
  __syncthreads();
  const float rstd = s_rstd;

  float* orow = out + (size_t)row * D_MODEL;
  for (int j = threadIdx.x; j < D_MODEL; j += 256)
    orow[j] = (xr[j] - mu) * rstd * gamma[j] + beta[j];
}

// ---------------------------------------------------------------------------
// Launch
// ---------------------------------------------------------------------------
extern "C" void kernel_launch(void* const* d_in, const int* in_sizes, int n_in,
                              void* d_out, int out_size, void* d_ws, size_t ws_size,
                              hipStream_t stream) {
  const float* q  = (const float*)d_in[0];
  const float* k  = (const float*)d_in[1];
  const float* v  = (const float*)d_in[2];
  const float* Wq = (const float*)d_in[3];
  const float* Wk = (const float*)d_in[4];
  const float* Wv = (const float*)d_in[5];
  const float* Wo = (const float*)d_in[6];
  const float* bo = (const float*)d_in[7];
  const float* g  = (const float*)d_in[8];
  const float* be = (const float*)d_in[9];

  // Workspace layout (88 MB total)
  char* ws = (char*)d_ws;
  const size_t MB = 1024u * 1024u;
  _Float16* qh  = (_Float16*)(ws + 0 * MB);   // [BS, D] f16
  _Float16* kh  = (_Float16*)(ws + 8 * MB);
  _Float16* vh  = (_Float16*)(ws + 16 * MB);
  _Float16* wqh = (_Float16*)(ws + 24 * MB);  // [D, D] f16
  _Float16* wkh = (_Float16*)(ws + 26 * MB);
  _Float16* wvh = (_Float16*)(ws + 28 * MB);
  _Float16* woh = (_Float16*)(ws + 30 * MB);
  _Float16* Qp  = (_Float16*)(ws + 32 * MB);  // [B,H,S,DK] f16
  _Float16* Kp  = (_Float16*)(ws + 40 * MB);  // [B,H,S,DK] f16
  float*    Vp  = (float*)   (ws + 48 * MB);  // [B,H,S,DK] f32
  _Float16* ao  = (_Float16*)(ws + 64 * MB);  // [B,S,D]   f16
  float*    tmp = (float*)   (ws + 72 * MB);  // [B,S,D]   f32

  const int NACT = BS * D_MODEL;       // 4M
  const int NW   = D_MODEL * D_MODEL;  // 1M

  k_f32_to_f16<<<256, 256, 0, stream>>>(q,  qh,  NACT);
  k_f32_to_f16<<<256, 256, 0, stream>>>(k,  kh,  NACT);
  k_f32_to_f16<<<256, 256, 0, stream>>>(v,  vh,  NACT);
  k_f32_to_f16<<<128, 256, 0, stream>>>(Wq, wqh, NW);
  k_f32_to_f16<<<128, 256, 0, stream>>>(Wk, wkh, NW);
  k_f32_to_f16<<<128, 256, 0, stream>>>(Wv, wvh, NW);
  k_f32_to_f16<<<128, 256, 0, stream>>>(Wo, woh, NW);

  dim3 ggrid(D_MODEL / BN, BS / BM);   // (8, 32) = 256 blocks
  k_gemm_xwT<<<ggrid, 256, 0, stream>>>(qh, wqh, Qp, nullptr, nullptr, nullptr, 0);
  k_gemm_xwT<<<ggrid, 256, 0, stream>>>(kh, wkh, Kp, nullptr, nullptr, nullptr, 0);
  k_gemm_xwT<<<ggrid, 256, 0, stream>>>(vh, wvh, nullptr, Vp, nullptr, nullptr, 1);

  float* out_main = (float*)d_out;                       // [B,S,D]
  float* out_sw   = out_main + (size_t)BS * D_MODEL;     // [B,H,S,TOPK]

  k_scores_topk<<<BATCH * NUM_HEADS * (SEQ / 16), 64, 0, stream>>>(
      Qp, Kp, Vp, ao, out_sw);

  k_gemm_xwT<<<ggrid, 256, 0, stream>>>(ao, woh, nullptr, tmp, bo, q, 2);
  k_layernorm<<<BS, 256, 0, stream>>>(tmp, g, be, out_main);
}
